// NonLocalBlock_88708254532072
// MI455X (gfx1250) — matrix operand order repaired and checked by reading, not compiled, not run on hardware
//
#include <hip/hip_runtime.h>
#include <hip/hip_bf16.h>

typedef __bf16 bf16_t;
typedef __attribute__((ext_vector_type(16))) __bf16 v16bf;
typedef __attribute__((ext_vector_type(8)))  __bf16 bf16x8;
typedef __attribute__((ext_vector_type(8)))  float  v8f;
typedef int v4i_vs __attribute__((vector_size(16)));   // matches builtin param type

static constexpr int BB = 4;     // batch
static constexpr int CC = 256;   // channels = head dim D
static constexpr int NN = 4096;  // positions
static constexpr int LDK = 264;  // K-stage row stride (528B, 16B multiple)
static constexpr int LDP = 40;   // P-stage row stride (80B, 16B multiple)

// ---------------- async copy global->LDS (gfx1250 path, guarded) -----------
#if defined(__has_builtin)
# if __has_builtin(__builtin_amdgcn_global_load_async_to_lds_b128)
#  define HAVE_ASYNC 1
# endif
#endif
#ifndef HAVE_ASYNC
# define HAVE_ASYNC 0
#endif

static __device__ __forceinline__ void cp16(const bf16_t* g, bf16_t* l) {
#if HAVE_ASYNC
    __builtin_amdgcn_global_load_async_to_lds_b128(
        (__attribute__((address_space(1))) v4i_vs*)g,
        (__attribute__((address_space(3))) v4i_vs*)l, 0, 0);
#else
    *(int4*)l = *(const int4*)g;          // sync fallback: b128 load + ds store
#endif
}

// ---- fragment loaders (contiguous k order: elem e <-> K = 16*half + e, wave32) ----

// 16 contiguous bf16 -> B-fragment lane payload (element e <-> K = e + 16*half)
static __device__ __forceinline__ v16bf ld16(const bf16_t* p) {
    bf16x8 lo = *(const bf16x8*)p;
    bf16x8 hi = *(const bf16x8*)(p + 8);
    return __builtin_shufflevector(lo, hi, 0,1,2,3,4,5,6,7,8,9,10,11,12,13,14,15);
}

// A 16x32 bf16 fragment from row-major base (row stride ld elements), K chunk at kbase.
// lanes 0-15: elems0-15 = K 0..15 ; lanes 16-31: K 16..31
static __device__ __forceinline__ v16bf lda(const bf16_t* base, int ld, int kbase,
                                            int m15, int half) {
    const bf16_t* row = base + (size_t)m15 * ld;
    bf16x8 lo = *(const bf16x8*)(row + kbase + half * 16);
    bf16x8 hi = *(const bf16x8*)(row + kbase + half * 16 + 8);
    return __builtin_shufflevector(lo, hi, 0,1,2,3,4,5,6,7,8,9,10,11,12,13,14,15);
}

static __device__ __forceinline__ v8f wmma_bf16(v16bf a, v16bf b, v8f c) {
    return __builtin_amdgcn_wmma_f32_16x16x32_bf16(false, a, false, b,
                                                   (short)0, c, false, false);
}

// ------------------------------------------------------------------
// Kernel 0: convert fp32 weights to bf16 in workspace
// ------------------------------------------------------------------
__global__ __launch_bounds__(256) void nlb_cvt(const float* __restrict__ wp,
                                               const float* __restrict__ wt,
                                               const float* __restrict__ wg,
                                               bf16_t* __restrict__ wk,
                                               bf16_t* __restrict__ wq,
                                               bf16_t* __restrict__ wv) {
    int i = blockIdx.x * 256 + threadIdx.x;
    if (i < CC * CC) {
        wk[i] = (bf16_t)wp[i];
        wq[i] = (bf16_t)wt[i];
        wv[i] = (bf16_t)wg[i];
    }
}

// ------------------------------------------------------------------
// Kernel 1: projections.  Q=theta [B,N,D], K=phi [B,N,D], Vt=g^T [B,D,N], bf16.
// ------------------------------------------------------------------
__global__ __launch_bounds__(128) void nlb_proj(const float* __restrict__ x,
                                                const bf16_t* __restrict__ wqb,
                                                const bf16_t* __restrict__ wkb,
                                                const bf16_t* __restrict__ wvb,
                                                const float* __restrict__ bq,
                                                const float* __restrict__ bk,
                                                const float* __restrict__ bv,
                                                bf16_t* __restrict__ Q,
                                                bf16_t* __restrict__ K,
                                                bf16_t* __restrict__ Vt) {
    constexpr int LDX = CC + 8;                 // 264: 528B row stride
    __shared__ bf16_t xT[16 * LDX];             // x tile transposed [n][c], bf16

    const int tid   = threadIdx.x;
    const int b     = blockIdx.x >> 8;          // 256 n-tiles per batch
    const int nt    = blockIdx.x & 255;
    const int nbase = nt * 16;

    for (int idx = tid; idx < CC * 16; idx += 128) {
        int c = idx >> 4, n = idx & 15;
        float v = x[((size_t)(b * CC + c)) * NN + nbase + n];
        xT[n * LDX + c] = (bf16_t)v;
    }
    __syncthreads();

    const int w    = tid >> 5;
    const int lane = tid & 31;
    const int m15  = lane & 15;
    const int half = (lane >> 4) & 1;

    const bf16_t* W[3]    = {wqb, wkb, wvb};
    const float*  bias[3] = {bq, bk, bv};

    v8f acc[3][4];
    for (int p = 0; p < 3; ++p)
        for (int j = 0; j < 4; ++j)
            for (int r = 0; r < 8; ++r) acc[p][j][r] = 0.0f;

    for (int kc = 0; kc < 8; ++kc) {
        v16bf a = lda(xT, LDX, kc * 32, m15, half);
        for (int p = 0; p < 3; ++p) {
            for (int j = 0; j < 4; ++j) {
                int dbase = (w + j * 4) * 16;
                v16bf bf = ld16(W[p] + (size_t)(dbase + m15) * CC + kc * 32 + half * 16);
                acc[p][j] = wmma_bf16(a, bf, acc[p][j]);
            }
        }
    }

    for (int p = 0; p < 3; ++p) {
        for (int j = 0; j < 4; ++j) {
            int d  = (w + j * 4) * 16 + m15;
            float bvv = bias[p][d];
            for (int r = 0; r < 8; ++r) {
                int   n   = nbase + r + half * 8;
                float val = acc[p][j][r] + bvv;
                if (p == 0)
                    Q[((size_t)(b * NN + n)) * CC + d] = (bf16_t)val;
                else if (p == 1)
                    K[((size_t)(b * NN + n)) * CC + d] = (bf16_t)val;
                else
                    Vt[((size_t)(b * CC + d)) * NN + n] = (bf16_t)val;
            }
        }
    }
}

// ------------------------------------------------------------------
// Kernel 2: flash attention + residual, double-buffered async K staging.
// Block = 128 threads (4 waves) = 2 q-subtiles (16 rows) x 2 d-halves (128 each).
// ------------------------------------------------------------------

// stage one 32-key x 256-d bf16 K chunk into LDS (8 x b128 per thread)
static __device__ __forceinline__ void stage_k(const bf16_t* kb_base, int kbase,
                                               bf16_t* dst, int tid) {
    for (int it = 0; it < 8; ++it) {
        int t    = tid + it * 128;       // 0..1023
        int row  = t >> 5;               // key 0..31
        int col8 = t & 31;               // 8-elem column group
        cp16(kb_base + (size_t)(kbase + row) * CC + col8 * 8,
             dst + row * LDK + col8 * 8);
    }
}

__global__ __launch_bounds__(128) void nlb_attn(const bf16_t* __restrict__ Q,
                                                const bf16_t* __restrict__ K,
                                                const bf16_t* __restrict__ Vt,
                                                const float* __restrict__ x,
                                                float* __restrict__ out) {
    __shared__ bf16_t kstage[2][32 * LDK];   // 2 x 16.9 KB double buffer
    __shared__ bf16_t plds[4 * 16 * LDP];    // per-wave P staging (C->A relayout)

    const int tid    = threadIdx.x;
    const int b      = blockIdx.x >> 7;      // 128 blocks per batch (32 q rows each)
    const int nt32   = blockIdx.x & 127;
    const int w      = tid >> 5;
    const int lane   = tid & 31;
    const int m15    = lane & 15;
    const int half   = (lane >> 4) & 1;
    const int qsub   = w & 1;
    const int dhalf  = w >> 1;
    const int qbase  = nt32 * 32 + qsub * 16;
    const int dbase0 = dhalf * 128;

    bf16_t* pw = plds + w * 16 * LDP;
    const bf16_t* kb_base = K + ((size_t)b * NN) * CC;

    // Q A-fragments: 8 chunks of the D=256 reduction
    const bf16_t* qrow = Q + ((size_t)(b * NN + qbase)) * CC;
    v16bf qa[8];
    for (int kc = 0; kc < 8; ++kc) qa[kc] = lda(qrow, CC, kc * 32, m15, half);

    v8f acc[8];
    float mrow[8], lrow[8], alpha[8];
    for (int j = 0; j < 8; ++j)
        for (int r = 0; r < 8; ++r) acc[j][r] = 0.0f;
    for (int r = 0; r < 8; ++r) { mrow[r] = -3.0e38f; lrow[r] = 0.0f; }

    stage_k(kb_base, 0, kstage[0], tid);     // prologue: chunk 0 -> buffer 0

    for (int kb = 0; kb < NN / 32; ++kb) {
        const int kbase = kb * 32;
        bf16_t* kcur = kstage[kb & 1];
        bf16_t* knxt = kstage[(kb & 1) ^ 1];
        const bool more = (kb + 1 < NN / 32);

        if (more) stage_k(kb_base, kbase + 32, knxt, tid);  // issue next chunk
#if HAVE_ASYNC
        // async loads complete in order: <=8 outstanding => current chunk landed
        if (more) asm volatile("s_wait_asynccnt 8" ::: "memory");
        else      asm volatile("s_wait_asynccnt 0" ::: "memory");
#endif
        __syncthreads();

        // ---- S = Q K^T : two 16x16 tiles from LDS-staged K ----
        v8f s0, s1;
        for (int r = 0; r < 8; ++r) { s0[r] = 0.0f; s1[r] = 0.0f; }
        for (int kc = 0; kc < 8; ++kc) {
            v16bf b0 = ld16(kcur + m15 * LDK        + kc * 32 + half * 16);
            v16bf b1 = ld16(kcur + (16 + m15) * LDK + kc * 32 + half * 16);
            s0 = wmma_bf16(qa[kc], b0, s0);
            s1 = wmma_bf16(qa[kc], b1, s1);
        }

        // ---- online softmax (row = r + 8*half, 16-lane group reductions) ----
        for (int r = 0; r < 8; ++r) {
            float mx = fmaxf(s0[r], s1[r]);
            for (int off = 8; off >= 1; off >>= 1)
                mx = fmaxf(mx, __shfl_xor(mx, off, 16));
            float mnew = fmaxf(mrow[r], mx);
            alpha[r] = __expf(mrow[r] - mnew);
            float e0 = __expf(s0[r] - mnew);
            float e1 = __expf(s1[r] - mnew);
            float sum = e0 + e1;
            for (int off = 8; off >= 1; off >>= 1)
                sum += __shfl_xor(sum, off, 16);
            lrow[r] = lrow[r] * alpha[r] + sum;
            mrow[r] = mnew;
            pw[(r + half * 8) * LDP + m15]      = (bf16_t)e0;
            pw[(r + half * 8) * LDP + 16 + m15] = (bf16_t)e1;
        }
        for (int j = 0; j < 8; ++j)
            for (int r = 0; r < 8; ++r) acc[j][r] *= alpha[r];

        asm volatile("s_wait_dscnt 0" ::: "memory");        // P staging RAW
        v16bf pA = lda(pw, LDP, 0, m15, half);              // A: M=q, K=key_local

        // ---- acc += P * V : B[K=key_local][N=d] = Vt[d][kbase+key_local] ----
        for (int j = 0; j < 8; ++j) {
            int d = dbase0 + j * 16 + m15;
            const bf16_t* vp = Vt + ((size_t)(b * CC + d)) * NN + kbase + half * 16;
            v16bf vb = ld16(vp);
            if (more) __builtin_prefetch(vp + 32, 0, 3);    // next chunk's V row
            acc[j] = wmma_bf16(pA, vb, acc[j]);
        }

        // all ds reads of kcur must land before it is recycled next iteration
        asm volatile("s_wait_dscnt 0" ::: "memory");
        __syncthreads();
    }

    // ---- epilogue: out[b][d][n] = acc/l + x[b][d][n] ----
    float rl[8];
    for (int r = 0; r < 8; ++r) rl[r] = 1.0f / lrow[r];
    for (int j = 0; j < 8; ++j) {
        int d = dbase0 + j * 16 + m15;
        size_t base = ((size_t)(b * CC + d)) * NN + qbase + half * 8;
        float4 x0 = *(const float4*)(x + base);
        float4 x1 = *(const float4*)(x + base + 4);
        float4 o0, o1;
        o0.x = acc[j][0] * rl[0] + x0.x;  o0.y = acc[j][1] * rl[1] + x0.y;
        o0.z = acc[j][2] * rl[2] + x0.z;  o0.w = acc[j][3] * rl[3] + x0.w;
        o1.x = acc[j][4] * rl[4] + x1.x;  o1.y = acc[j][5] * rl[5] + x1.y;
        o1.z = acc[j][6] * rl[6] + x1.z;  o1.w = acc[j][7] * rl[7] + x1.w;
        *(float4*)(out + base)     = o0;
        *(float4*)(out + base + 4) = o1;
    }
}

// ------------------------------------------------------------------
extern "C" void kernel_launch(void* const* d_in, const int* in_sizes, int n_in,
                              void* d_out, int out_size, void* d_ws, size_t ws_size,
                              hipStream_t stream) {
    const float* x   = (const float*)d_in[0];
    const float* wph = (const float*)d_in[1];  // phi   -> K
    const float* bph = (const float*)d_in[2];
    const float* wth = (const float*)d_in[3];  // theta -> Q
    const float* bth = (const float*)d_in[4];
    const float* wg  = (const float*)d_in[5];  // g     -> V
    const float* bg  = (const float*)d_in[6];
    float* out = (float*)d_out;

    char* ws = (char*)d_ws;
    const size_t qkvBytes = (size_t)BB * NN * CC * sizeof(bf16_t);  // 8 MB each
    bf16_t* Q  = (bf16_t*)(ws);
    bf16_t* K  = (bf16_t*)(ws + qkvBytes);
    bf16_t* Vt = (bf16_t*)(ws + 2 * qkvBytes);
    const size_t wBytes = (size_t)CC * CC * sizeof(bf16_t);         // 128 KB each
    bf16_t* wqb = (bf16_t*)(ws + 3 * qkvBytes);
    bf16_t* wkb = (bf16_t*)(ws + 3 * qkvBytes + wBytes);
    bf16_t* wvb = (bf16_t*)(ws + 3 * qkvBytes + 2 * wBytes);

    nlb_cvt<<<(CC * CC + 255) / 256, 256, 0, stream>>>(wph, wth, wg, wkb, wqb, wvb);
    nlb_proj<<<BB * (NN / 16), 128, 0, stream>>>(x, wqb, wkb, wvb,
                                                 bth, bph, bg, Q, K, Vt);
    nlb_attn<<<BB * (NN / 32), 128, 0, stream>>>(Q, K, Vt, x, out);
}
